// MultiHeadedSelfAttention_16930761081314
// MI455X (gfx1250) — matrix operand order repaired and checked by reading, not compiled, not run on hardware
//
#include <hip/hip_runtime.h>
#include <cstdint>

// ---------------------------------------------------------------------------
// MI455X (gfx1250) fused multi-head self-attention, f16 WMMA pipeline with
// async-to-LDS double-buffered staging (CDNA5 ASYNCcnt path).
//   sizes: tgt_len=2048, DIM=1024, N_HEADS=16, HEAD_DIM=64
//   ws layout (f16): xh(2M) winh(3M) wouth(1M) q(2M) k(2M) vT(2M) ctxh(2M)
//   = 14M halves = 28 MB scratch
// ---------------------------------------------------------------------------

typedef _Float16 f16;
typedef __attribute__((ext_vector_type(16))) _Float16 v16h;
typedef __attribute__((ext_vector_type(8)))  _Float16 v8h;
typedef __attribute__((ext_vector_type(4)))  _Float16 v4h;
typedef __attribute__((ext_vector_type(8)))  float    v8f;

#define TGT   2048
#define DIM   1024
#define NH    16
#define HD    64

union FragU { v16h v; v8h h[2]; };

__device__ __forceinline__ v8f wmma16(v16h a, v16h b, v8f c) {
  // v_wmma_f32_16x16x32_f16  (D = A(16x32) * B(32x16) + C)
  return __builtin_amdgcn_wmma_f32_16x16x32_f16(false, a, false, b,
                                                (short)0, c, false, false);
}

// ---- CDNA5 async copy: global -> LDS, 16B per lane, ASYNCcnt-tracked ------
__device__ __forceinline__ void async_b128(uint32_t lds_byte_off,
                                           const void* gptr) {
  asm volatile("global_load_async_to_lds_b128 %0, %1, off"
               :: "v"(lds_byte_off), "v"((uint64_t)(uintptr_t)gptr)
               : "memory");
}
__device__ __forceinline__ void wait_async0() {
  asm volatile("s_wait_asynccnt 0x0" ::: "memory");
}
__device__ __forceinline__ void wait_async2() {
  asm volatile("s_wait_asynccnt 0x2" ::: "memory");
}
__device__ __forceinline__ void wait_async4() {
  asm volatile("s_wait_asynccnt 0x4" ::: "memory");
}
// low 32 bits of a generic pointer to __shared__ == LDS byte address
__device__ __forceinline__ uint32_t lds_off(const void* p) {
  return (uint32_t)(uintptr_t)p;
}

// A fragment (16x32, MxK) from row-major [*, ld] f16 at (row0, k0).
// ISA layout: lane<16 -> row=lane, K {0..7,16..23}; lane>=16 -> K {8..15,24..31}
__device__ __forceinline__ v16h load_frag_a(const f16* base, int ld,
                                            int row0, int k0, int lane) {
  const int m = lane & 15, hf = lane >> 4;
  const f16* p = base + (size_t)(row0 + m) * ld + k0 + hf * 8;
  FragU u;
  u.h[0] = *(const v8h*)(p);
  u.h[1] = *(const v8h*)(p + 16);
  return u.v;
}

// B fragment (32x16, KxN) where B[k][n] = M[col0+n][k0+k] for row-major M[*,ld].
// ISA layout: lane<16 -> col=lane, K {0..7,16..23}; lane>=16 -> K {8..15,24..31}
__device__ __forceinline__ v16h load_frag_b(const f16* base, int ld,
                                            int col0, int k0, int lane) {
  const int n = lane & 15, kh = lane >> 4;
  const f16* p = base + (size_t)(col0 + n) * ld + k0 + kh * 8;
  FragU u;
  u.h[0] = *(const v8h*)(p);
  u.h[1] = *(const v8h*)(p + 16);
  return u.v;
}

// ---------------------------------------------------------------------------
// f32 -> f16 conversion (vectorized x4)
// ---------------------------------------------------------------------------
__global__ void cvt_f32_f16_kernel(const float* __restrict__ src,
                                   f16* __restrict__ dst, int n4) {
  int i = blockIdx.x * blockDim.x + threadIdx.x;
  if (i < n4) {
    float4 v = ((const float4*)src)[i];
    v4h o;
    o[0] = (f16)v.x; o[1] = (f16)v.y; o[2] = (f16)v.z; o[3] = (f16)v.w;
    ((v4h*)dst)[i] = o;
  }
}

// ---------------------------------------------------------------------------
// GEMM1: qkv = xh[2048,1024] @ winh[3072,1024]^T, scattered into
//   q[h][n][d], k[h][n][d], vT[h][d][n]  (all f16)
// 8 waves/WG, wave tile 32x64, WG tile 128x128, async double-buffered LDS.
// grid (16, 24).
// ---------------------------------------------------------------------------
__global__ __launch_bounds__(256)
void gemm_qkv_kernel(const f16* __restrict__ xh, const f16* __restrict__ wh,
                     f16* __restrict__ q, f16* __restrict__ kk,
                     f16* __restrict__ vT) {
  __shared__ __align__(16) f16 sA[2][128 * 32];   // [row][k] 64B rows
  __shared__ __align__(16) f16 sB[2][128 * 32];   // [col][k] 64B rows
  const int tid  = threadIdx.x;
  const int lane = tid & 31;
  const int wave = tid >> 5;
  const int M0 = blockIdx.x * 128;
  const int N0 = blockIdx.y * 128;
  const int mw = (wave >> 1) * 32;
  const int nw = (wave & 1) * 64;
  const int cr = tid >> 2;      // 0..63 : tile row
  const int cc = tid & 3;       // 0..3  : 16B chunk within 64B row

  auto issue_tile = [&](int buf, int k0) {
    uint32_t la = lds_off(&sA[buf][0]);
    uint32_t lb = lds_off(&sB[buf][0]);
    const f16* gA = xh + (size_t)(M0 + cr) * DIM + k0 + cc * 8;
    const f16* gB = wh + (size_t)(N0 + cr) * DIM + k0 + cc * 8;
    async_b128(la + cr * 64 + cc * 16,        gA);
    async_b128(la + (cr + 64) * 64 + cc * 16, gA + (size_t)64 * DIM);
    async_b128(lb + cr * 64 + cc * 16,        gB);
    async_b128(lb + (cr + 64) * 64 + cc * 16, gB + (size_t)64 * DIM);
  };

  const v8f zero = {0.f, 0.f, 0.f, 0.f, 0.f, 0.f, 0.f, 0.f};
  v8f acc[2][4];
#pragma unroll
  for (int i = 0; i < 2; ++i)
#pragma unroll
    for (int j = 0; j < 4; ++j) acc[i][j] = zero;

  issue_tile(0, 0);
  issue_tile(1, 32);

  const int NT = DIM / 32;   // 32 k-steps
  for (int i = 0; i < NT; ++i) {
    if (i + 1 < NT) wait_async4(); else wait_async0();
    __syncthreads();
    const f16* A = &sA[i & 1][0];
    const f16* B = &sB[i & 1][0];
    v16h a0 = load_frag_a(A, 32, mw,      0, lane);
    v16h a1 = load_frag_a(A, 32, mw + 16, 0, lane);
    v16h b0 = load_frag_b(B, 32, nw +  0, 0, lane);
    v16h b1 = load_frag_b(B, 32, nw + 16, 0, lane);
    v16h b2 = load_frag_b(B, 32, nw + 32, 0, lane);
    v16h b3 = load_frag_b(B, 32, nw + 48, 0, lane);
    acc[0][0] = wmma16(a0, b0, acc[0][0]);
    acc[0][1] = wmma16(a0, b1, acc[0][1]);
    acc[0][2] = wmma16(a0, b2, acc[0][2]);
    acc[0][3] = wmma16(a0, b3, acc[0][3]);
    acc[1][0] = wmma16(a1, b0, acc[1][0]);
    acc[1][1] = wmma16(a1, b1, acc[1][1]);
    acc[1][2] = wmma16(a1, b2, acc[1][2]);
    acc[1][3] = wmma16(a1, b3, acc[1][3]);
    __syncthreads();
    if (i + 2 < NT) issue_tile(i & 1, (i + 2) * 32);
  }

  // C layout: VGPR r -> row (r or r+8 by lane half), lane&15 -> col
  const int colb = lane & 15, rowoff = (lane >> 4) * 8;
#pragma unroll
  for (int i = 0; i < 2; ++i)
#pragma unroll
    for (int j = 0; j < 4; ++j)
#pragma unroll
      for (int r = 0; r < 8; ++r) {
        int row = M0 + mw + i * 16 + rowoff + r;
        int col = N0 + nw + j * 16 + colb;
        f16 val = (f16)acc[i][j][r];
        int sec = col >> 10;          // 0:q 1:k 2:v
        int wi  = col & 1023;
        int h   = wi >> 6;
        int d   = wi & 63;
        if (sec == 0)      q [((size_t)(h * TGT + row)) * HD + d] = val;
        else if (sec == 1) kk[((size_t)(h * TGT + row)) * HD + d] = val;
        else               vT[((size_t)(h * HD + d)) * TGT + row] = val;
      }
}

// ---------------------------------------------------------------------------
// Fused flash attention per head. 8 waves/WG, each wave = 16 query rows.
// K/V 32-key tiles async-staged into LDS, shared by all 8 waves (removes the
// 8x duplicate global fetch). grid (TGT/128 = 16, NH = 16).
// ---------------------------------------------------------------------------
__global__ __launch_bounds__(256)
void attn_kernel(const f16* __restrict__ q, const f16* __restrict__ kk,
                 const f16* __restrict__ vT, f16* __restrict__ ctxh,
                 float scaling) {
  __shared__ __align__(16) f16 sK[2][32 * 64];    // [key][dim]   128B rows
  __shared__ __align__(16) f16 sV[2][64 * 32];    // [dim][key]    64B rows
  __shared__ __align__(16) f16 sP[8][16 * 32];    // per-wave P transpose slab
  const int tid  = threadIdx.x;
  const int lane = tid & 31;
  const int wave = tid >> 5;
  const int h = blockIdx.y;
  const int row0 = blockIdx.x * 128 + wave * 16;

  const f16* qh = q  + (size_t)h * TGT * HD;
  const f16* kh = kk + (size_t)h * TGT * HD;
  const f16* vh = vT + (size_t)h * HD * TGT;    // [HD][TGT]

  // cooperative K/V tile copy mapping (16B per thread per instruction)
  const int krow = tid >> 3, kchk = tid & 7;    // K: 32 rows x 8 chunks
  const int vrow = tid >> 2, vchk = tid & 3;    // V: 64 rows x 4 chunks
  auto issue_kv = [&](int buf, int kt) {
    async_b128(lds_off(&sK[buf][0]) + krow * 128 + kchk * 16,
               kh + (size_t)(kt + krow) * HD + kchk * 8);
    async_b128(lds_off(&sV[buf][0]) + vrow * 64 + vchk * 16,
               vh + (size_t)vrow * TGT + kt + vchk * 8);
  };

  // 16 rows x K=64 of Q kept resident
  v16h qa0 = load_frag_a(qh, HD, row0,  0, lane);
  v16h qa1 = load_frag_a(qh, HD, row0, 32, lane);

  const v8f zero = {0.f, 0.f, 0.f, 0.f, 0.f, 0.f, 0.f, 0.f};
  v8f acc[4];
#pragma unroll
  for (int j = 0; j < 4; ++j) acc[j] = zero;

  float mr[8], lr[8];
#pragma unroll
  for (int r = 0; r < 8; ++r) { mr[r] = -1e30f; lr[r] = 0.f; }

  f16* myp = &sP[wave][0];
  const int ml = lane & 15, hf = lane >> 4;

  issue_kv(0, 0);
  issue_kv(1, 32);

  const int NT = TGT / 32;   // 64 key tiles
  for (int i = 0; i < NT; ++i) {
    if (i + 1 < NT) wait_async2(); else wait_async0();
    __syncthreads();
    const f16* K = &sK[i & 1][0];
    const f16* V = &sV[i & 1][0];

    // ---- S = (Q @ K^T) * scaling, 16 x 32 tile ----
    v16h kb00 = load_frag_b(K, HD,  0,  0, lane);
    v16h kb01 = load_frag_b(K, HD,  0, 32, lane);
    v16h kb10 = load_frag_b(K, HD, 16,  0, lane);
    v16h kb11 = load_frag_b(K, HD, 16, 32, lane);
    v8f s0 = zero, s1 = zero;
    s0 = wmma16(qa0, kb00, s0);
    s0 = wmma16(qa1, kb01, s0);
    s1 = wmma16(qa0, kb10, s1);
    s1 = wmma16(qa1, kb11, s1);

    float mt[8], ps[8];
#pragma unroll
    for (int r = 0; r < 8; ++r) {
      s0[r] *= scaling;
      s1[r] *= scaling;
      mt[r] = fmaxf(s0[r], s1[r]);
    }
    // row max across the 16 lanes holding this row's 16 columns
#pragma unroll
    for (int off = 8; off; off >>= 1)
#pragma unroll
      for (int r = 0; r < 8; ++r)
        mt[r] = fmaxf(mt[r], __shfl_xor(mt[r], off, 32));

    float corr[8];
#pragma unroll
    for (int r = 0; r < 8; ++r) {
      float mn = fmaxf(mr[r], mt[r]);
      corr[r] = __expf(mr[r] - mn);
      mr[r] = mn;
      float p0 = __expf(s0[r] - mn);
      float p1 = __expf(s1[r] - mn);
      // C-layout -> LDS (row, key); rows r / r+8 by lane half
      myp[(r + hf * 8) * 32 + ml]      = (f16)p0;
      myp[(r + hf * 8) * 32 + 16 + ml] = (f16)p1;
      ps[r] = p0 + p1;
    }
#pragma unroll
    for (int off = 8; off; off >>= 1)
#pragma unroll
      for (int r = 0; r < 8; ++r)
        ps[r] += __shfl_xor(ps[r], off, 32);
#pragma unroll
    for (int r = 0; r < 8; ++r) lr[r] = lr[r] * corr[r] + ps[r];
#pragma unroll
    for (int j = 0; j < 4; ++j)
#pragma unroll
      for (int r = 0; r < 8; ++r) acc[j][r] *= corr[r];

    // ---- ctx += P(16x32) @ V(32x64); A from LDS transpose, B from sV ----
    v16h pa  = load_frag_a(myp, 32, 0, 0, lane);
    v16h vb0 = load_frag_b(V, 32,  0, 0, lane);
    v16h vb1 = load_frag_b(V, 32, 16, 0, lane);
    v16h vb2 = load_frag_b(V, 32, 32, 0, lane);
    v16h vb3 = load_frag_b(V, 32, 48, 0, lane);
    acc[0] = wmma16(pa, vb0, acc[0]);
    acc[1] = wmma16(pa, vb1, acc[1]);
    acc[2] = wmma16(pa, vb2, acc[2]);
    acc[3] = wmma16(pa, vb3, acc[3]);

    __syncthreads();
    if (i + 2 < NT) issue_kv(i & 1, (i + 2) * 32);
  }

  // normalize + store ctx as f16 [row][h*64+d]
#pragma unroll
  for (int j = 0; j < 4; ++j)
#pragma unroll
    for (int r = 0; r < 8; ++r) {
      float o = acc[j][r] / lr[r];
      int row = row0 + hf * 8 + r;
      int col = h * HD + j * 16 + ml;
      ctxh[(size_t)row * DIM + col] = (f16)o;
    }
}

// ---------------------------------------------------------------------------
// GEMM3: out[2048,1024] (fp32) = ctxh[2048,1024] @ wouth[1024,1024]^T
// Same async-staged tiling as GEMM1. grid (16, 8).
// ---------------------------------------------------------------------------
__global__ __launch_bounds__(256)
void gemm_out_kernel(const f16* __restrict__ ctxh, const f16* __restrict__ woh,
                     float* __restrict__ out) {
  __shared__ __align__(16) f16 sA[2][128 * 32];
  __shared__ __align__(16) f16 sB[2][128 * 32];
  const int tid  = threadIdx.x;
  const int lane = tid & 31;
  const int wave = tid >> 5;
  const int M0 = blockIdx.x * 128;
  const int N0 = blockIdx.y * 128;
  const int mw = (wave >> 1) * 32;
  const int nw = (wave & 1) * 64;
  const int cr = tid >> 2;
  const int cc = tid & 3;

  auto issue_tile = [&](int buf, int k0) {
    uint32_t la = lds_off(&sA[buf][0]);
    uint32_t lb = lds_off(&sB[buf][0]);
    const f16* gA = ctxh + (size_t)(M0 + cr) * DIM + k0 + cc * 8;
    const f16* gB = woh  + (size_t)(N0 + cr) * DIM + k0 + cc * 8;
    async_b128(la + cr * 64 + cc * 16,        gA);
    async_b128(la + (cr + 64) * 64 + cc * 16, gA + (size_t)64 * DIM);
    async_b128(lb + cr * 64 + cc * 16,        gB);
    async_b128(lb + (cr + 64) * 64 + cc * 16, gB + (size_t)64 * DIM);
  };

  const v8f zero = {0.f, 0.f, 0.f, 0.f, 0.f, 0.f, 0.f, 0.f};
  v8f acc[2][4];
#pragma unroll
  for (int i = 0; i < 2; ++i)
#pragma unroll
    for (int j = 0; j < 4; ++j) acc[i][j] = zero;

  issue_tile(0, 0);
  issue_tile(1, 32);

  const int NT = DIM / 32;
  for (int i = 0; i < NT; ++i) {
    if (i + 1 < NT) wait_async4(); else wait_async0();
    __syncthreads();
    const f16* A = &sA[i & 1][0];
    const f16* B = &sB[i & 1][0];
    v16h a0 = load_frag_a(A, 32, mw,      0, lane);
    v16h a1 = load_frag_a(A, 32, mw + 16, 0, lane);
    v16h b0 = load_frag_b(B, 32, nw +  0, 0, lane);
    v16h b1 = load_frag_b(B, 32, nw + 16, 0, lane);
    v16h b2 = load_frag_b(B, 32, nw + 32, 0, lane);
    v16h b3 = load_frag_b(B, 32, nw + 48, 0, lane);
    acc[0][0] = wmma16(a0, b0, acc[0][0]);
    acc[0][1] = wmma16(a0, b1, acc[0][1]);
    acc[0][2] = wmma16(a0, b2, acc[0][2]);
    acc[0][3] = wmma16(a0, b3, acc[0][3]);
    acc[1][0] = wmma16(a1, b0, acc[1][0]);
    acc[1][1] = wmma16(a1, b1, acc[1][1]);
    acc[1][2] = wmma16(a1, b2, acc[1][2]);
    acc[1][3] = wmma16(a1, b3, acc[1][3]);
    __syncthreads();
    if (i + 2 < NT) issue_tile(i & 1, (i + 2) * 32);
  }

  const int colb = lane & 15, rowoff = (lane >> 4) * 8;
#pragma unroll
  for (int i = 0; i < 2; ++i)
#pragma unroll
    for (int j = 0; j < 4; ++j)
#pragma unroll
      for (int r = 0; r < 8; ++r) {
        int row = M0 + mw + i * 16 + rowoff + r;
        int col = N0 + nw + j * 16 + colb;
        out[(size_t)row * DIM + col] = acc[i][j][r];
      }
}

// ---------------------------------------------------------------------------
extern "C" void kernel_launch(void* const* d_in, const int* in_sizes, int n_in,
                              void* d_out, int out_size, void* d_ws,
                              size_t ws_size, hipStream_t stream) {
  const float* x     = (const float*)d_in[0];   // [2048,1024]
  const float* W_in  = (const float*)d_in[1];   // [3072,1024]
  const float* W_out = (const float*)d_in[2];   // [1024,1024]
  float* out = (float*)d_out;

  f16* ws    = (f16*)d_ws;
  f16* xh    = ws;                                  // 2M halves
  f16* winh  = xh    + (size_t)2 * 1024 * 1024;     // 3M
  f16* wouth = winh  + (size_t)3 * 1024 * 1024;     // 1M
  f16* qb    = wouth + (size_t)1 * 1024 * 1024;     // 2M
  f16* kb    = qb    + (size_t)2 * 1024 * 1024;     // 2M
  f16* vTb   = kb    + (size_t)2 * 1024 * 1024;     // 2M
  f16* ctxh  = vTb   + (size_t)2 * 1024 * 1024;     // 2M  (total 28 MB)

  // quake fast-rsqrt(64) with one Newton step, rounded to fp16 (as reference):
  // y0 = 0.120777, y1 = 0.1247885 -> fp16 = 2045/16384
  const float scaling = 0.12481689453125f;

  {
    int n4 = TGT * DIM / 4;
    cvt_f32_f16_kernel<<<(n4 + 255) / 256, 256, 0, stream>>>(x, xh, n4);
  }
  {
    int n4 = 3 * DIM * DIM / 4;
    cvt_f32_f16_kernel<<<(n4 + 255) / 256, 256, 0, stream>>>(W_in, winh, n4);
  }
  {
    int n4 = DIM * DIM / 4;
    cvt_f32_f16_kernel<<<(n4 + 255) / 256, 256, 0, stream>>>(W_out, wouth, n4);
  }

  gemm_qkv_kernel<<<dim3(TGT / 128, 3 * DIM / 128), 256, 0, stream>>>(
      xh, winh, qb, kb, vTb);

  attn_kernel<<<dim3(TGT / 128, NH), 256, 0, stream>>>(
      qb, kb, vTb, ctxh, scaling);

  gemm_out_kernel<<<dim3(TGT / 128, DIM / 128), 256, 0, stream>>>(
      ctxh, wouth, out);
}
